// NeuronCircuit_21157008900535
// MI455X (gfx1250) — matrix operand order repaired and checked by reading, not compiled, not run on hardware
//
#include <hip/hip_runtime.h>

// ---------- problem constants ----------
#define BB 8
#define SS 1024
#define DD 1024
#define HH 16
#define DH 64
#define RR 128
#define KC 16
#define KE 8
#define SCALE 0.125f   // 1/sqrt(64)

typedef __bf16  v8bf  __attribute__((ext_vector_type(8)));
typedef __bf16  v16bf __attribute__((ext_vector_type(16)));
typedef float   v8f   __attribute__((ext_vector_type(8)));

static __device__ __forceinline__ v8f vzero() {
    v8f z;
    #pragma unroll
    for (int i = 0; i < 8; ++i) z[i] = 0.0f;
    return z;
}

// A fragment (16x32 bf16): lane holds row M=lane&15; elems 0..7 = K kb..kb+7,
// elems 8..15 = K kb+16..kb+23, kb = (lane>>4)*8.  Two 16B loads.
static __device__ __forceinline__ v16bf load_afrag(const __bf16* p) {
    v8bf lo = *(const v8bf*)p;
    v8bf hi = *(const v8bf*)(p + 16);
    v16bf r;
    #pragma unroll
    for (int i = 0; i < 8; ++i) { r[i] = lo[i]; r[i + 8] = hi[i]; }
    return r;
}
// B fragment (32x16 bf16) from B^T row-major: lane holds col N=lane&15,
// elems 0..7 = K kb..kb+7, elems 8..15 = K kb+16..kb+23, kb = (lane>>4)*8.  Two 16B loads.
static __device__ __forceinline__ v16bf load_bfrag(const __bf16* p) {
    v8bf lo = *(const v8bf*)p;
    v8bf hi = *(const v8bf*)(p + 16);
    v16bf r;
    #pragma unroll
    for (int i = 0; i < 8; ++i) { r[i] = lo[i]; r[i + 8] = hi[i]; }
    return r;
}

static __device__ __forceinline__ v8f wmma_bf16(v16bf a, v16bf b, v8f c) {
    return __builtin_amdgcn_wmma_f32_16x16x32_bf16(false, a, false, b, (short)0, c, false, false);
}

// ---------- small prep kernels ----------
__global__ void cvt_f32_bf16(const float* __restrict__ in, __bf16* __restrict__ out, int n) {
    int i = blockIdx.x * blockDim.x + threadIdx.x;
    if (i < n) out[i] = (__bf16)in[i];
}

// WcT[b][r][d] = sum_k w[b,k] * neurons[idx[b,k]][d][r]
__global__ void build_wct(const float* __restrict__ w, const int* __restrict__ idx,
                          const float* __restrict__ neurons, __bf16* __restrict__ out) {
    int i = blockIdx.x * blockDim.x + threadIdx.x;           // over B*R*D, d fastest
    if (i >= BB * RR * DD) return;
    int d = i % DD, r = (i / DD) % RR, b = i / (RR * DD);
    float acc = 0.f;
    #pragma unroll
    for (int k = 0; k < KC; ++k) {
        int n = idx[b * KC + k];
        acc += w[b * KC + k] * neurons[(size_t)n * DD * RR + (size_t)d * RR + r];
    }
    out[i] = (__bf16)acc;
}

// WeT[b][d][r] = sum_k w[b,k] * pool[idx[b,k]][r][d]
__global__ void build_wet(const float* __restrict__ w, const int* __restrict__ idx,
                          const float* __restrict__ pool, __bf16* __restrict__ out) {
    int i = blockIdx.x * blockDim.x + threadIdx.x;           // over B*D*R, r fastest
    if (i >= BB * DD * RR) return;
    int r = i % RR, d = (i / RR) % DD, b = i / (DD * RR);
    float acc = 0.f;
    #pragma unroll
    for (int k = 0; k < KE; ++k) {
        int n = idx[b * KE + k];
        acc += w[b * KE + k] * pool[(size_t)n * RR * DD + (size_t)r * DD + d];
    }
    out[i] = (__bf16)acc;
}

// ---------- generic batched WMMA GEMM:  C[M,N] = A[M,K] * B[K,N] ----------
// Wave tile = 32(M) x 64(N): 2 A frags share 4 B frags -> 8 WMMAs per k-step.
// KDIM/LDA/LDBT/LDC are compile-time: the k-loop fully unrolls, all loads
// become base+immediate, and fragments are SSA values (no ping-pong moves).
// MODE 0: bf16 store C[r*ldc+c]; 1: bf16 transposed C[c*ldc+r]; 2: f32 store.
template<int KDIM, int LDA, int LDBT, int LDC, int MODE>
__global__ void gemm_wmma(const __bf16* __restrict__ A, const __bf16* __restrict__ Bt,
                          void* __restrict__ C,
                          unsigned long long sA, unsigned long long sBt, unsigned long long sC) {
    const int wave = threadIdx.x >> 5, lane = threadIdx.x & 31;
    const int lq = lane & 15, hi = lane >> 4;
    const int m0 = blockIdx.y * 256 + wave * 32;
    const int n0 = blockIdx.x * 64;
    A  += (size_t)blockIdx.z * sA;
    Bt += (size_t)blockIdx.z * sBt;

    const __bf16* arow0 = A + (size_t)(m0 + lq) * LDA + hi * 8;
    const __bf16* arow1 = arow0 + (size_t)16 * LDA;
    const __bf16* brow[4];
    #pragma unroll
    for (int t = 0; t < 4; ++t)
        brow[t] = Bt + (size_t)(n0 + t * 16 + lq) * LDBT + hi * 8;

    v8f acc[2][4];
    #pragma unroll
    for (int s = 0; s < 2; ++s)
        #pragma unroll
        for (int t = 0; t < 4; ++t) acc[s][t] = vzero();

    #pragma unroll
    for (int k = 0; k < KDIM; k += 32) {
        if (k + 256 < KDIM) __builtin_prefetch(arow0 + k + 256, 0, 1);
        v16bf a0 = load_afrag(arow0 + k);
        v16bf a1 = load_afrag(arow1 + k);
        #pragma unroll
        for (int t = 0; t < 4; ++t) {
            v16bf bf = load_bfrag(brow[t] + k);
            acc[0][t] = wmma_bf16(a0, bf, acc[0][t]);
            acc[1][t] = wmma_bf16(a1, bf, acc[1][t]);
        }
    }
    // epilogue
    #pragma unroll
    for (int s = 0; s < 2; ++s) {
        #pragma unroll
        for (int t = 0; t < 4; ++t) {
            #pragma unroll
            for (int i = 0; i < 8; ++i) {
                int r = m0 + s * 16 + i + hi * 8;
                int c = n0 + t * 16 + lq;
                if constexpr (MODE == 0) {
                    ((__bf16*)C)[(size_t)blockIdx.z * sC + (size_t)r * LDC + c] = (__bf16)acc[s][t][i];
                } else if constexpr (MODE == 1) {
                    ((__bf16*)C)[(size_t)blockIdx.z * sC + (size_t)c * LDC + r] = (__bf16)acc[s][t][i];
                } else {
                    ((float*)C)[(size_t)blockIdx.z * sC + (size_t)r * LDC + c] = acc[s][t][i];
                }
            }
        }
    }
}

// ---------- flash attention (transposed-score trick, software-pipelined) ----------
// Q,K: [B,S,D] bf16 (D = H*DH);  VT: [B,H,DH,S] bf16;  O: [B,S,D] bf16.
__global__ void attn_wmma(const __bf16* __restrict__ Q, const __bf16* __restrict__ K,
                          const __bf16* __restrict__ VT, __bf16* __restrict__ O) {
    const int b = blockIdx.z, h = blockIdx.y;
    const int wave = threadIdx.x >> 5, lane = threadIdx.x & 31;
    const int lq = lane & 15, hi = lane >> 4;
    const int q0 = (blockIdx.x * 8 + wave) * 16;
    const int qa = q0 + lq;                      // absolute query owned by this lane
    const int qlast = q0 + 15;                   // last k-block start <= qlast

    // Q as B operand (contraction = d), two 32-wide d chunks
    const __bf16* qrow = Q + ((size_t)b * SS + q0 + lq) * DD + h * DH;
    v16bf qf[2];
    #pragma unroll
    for (int c = 0; c < 2; ++c) qf[c] = load_bfrag(qrow + c * 32 + hi * 8);

    const __bf16* kbase = K + ((size_t)b * SS + lq) * DD + h * DH + hi * 8;
    const __bf16* vbase = VT + (((size_t)b * HH + h) * DH + lq) * SS + hi * 8;

    v8f acc[4];
    #pragma unroll
    for (int t = 0; t < 4; ++t) acc[t] = vzero();
    float m_run = -__builtin_inff(), l_run = 0.f;

    // prologue: K fragments for kb = 0  (kf[t][c]: key tile t, d-chunk c)
    v16bf kf[2][2];
    #pragma unroll
    for (int t = 0; t < 2; ++t) {
        #pragma unroll
        for (int c = 0; c < 2; ++c)
            kf[t][c] = load_afrag(kbase + (size_t)(t * 16) * DD + c * 32);
    }

    for (int kb = 0; kb <= qlast; kb += 32) {
        // scores first: consume K fragments loaded one iteration ago
        v8f st[2];
        #pragma unroll
        for (int t = 0; t < 2; ++t) {
            st[t] = wmma_bf16(kf[t][0], qf[0], vzero());
            st[t] = wmma_bf16(kf[t][1], qf[1], st[t]);
        }
        // issue this block's VT loads and next block's K loads NOW,
        // so they overlap the softmax VALU below
        v16bf vf[4];
        #pragma unroll
        for (int n = 0; n < 4; ++n)
            vf[n] = load_bfrag(vbase + (size_t)(n * 16) * SS + kb);
        const int kbn = (kb + 32 <= qlast) ? (kb + 32) : kb;   // branch-free clamp
        v16bf nk[2][2];
        #pragma unroll
        for (int t = 0; t < 2; ++t) {
            #pragma unroll
            for (int c = 0; c < 2; ++c)
                nk[t][c] = load_afrag(kbase + (size_t)(kbn + t * 16) * DD + c * 32);
        }
        // per-lane softmax over 16 key-scores for query qa
        float vals[16]; float tmax = -__builtin_inff();
        #pragma unroll
        for (int t = 0; t < 2; ++t) {
            #pragma unroll
            for (int i = 0; i < 8; ++i) {
                int kk = kb + t * 16 + hi * 8 + i;
                float v = st[t][i] * SCALE;
                if (kk > qa) v = -__builtin_inff();
                vals[t * 8 + i] = v;
                tmax = fmaxf(tmax, v);
            }
        }
        tmax = fmaxf(tmax, __shfl_xor(tmax, 16, 32));     // merge half-rows (same q)
        float m_new = fmaxf(m_run, tmax);
        float alpha = __expf(m_run - m_new);
        float p[16], psum = 0.f;
        #pragma unroll
        for (int j = 0; j < 16; ++j) { p[j] = __expf(vals[j] - m_new); psum += p[j]; }
        psum += __shfl_xor(psum, 16, 32);
        l_run = l_run * alpha + psum;
        m_run = m_new;
        // rescale accumulators: alpha for row (q0+i+hi*8) lives in lane (i+hi*8)
        #pragma unroll
        for (int i = 0; i < 8; ++i) {
            float a = __shfl(alpha, i + hi * 8, 32);
            #pragma unroll
            for (int t = 0; t < 4; ++t) acc[t][i] *= a;
        }
        // P as A operand: per-lane C-layout of S^T is already A-layout of P
        v16bf pf;
        #pragma unroll
        for (int i = 0; i < 8; ++i) { pf[i] = (__bf16)p[i]; pf[i + 8] = (__bf16)p[8 + i]; }
        // O += P * V  (vf issued at top of this iteration)
        #pragma unroll
        for (int n = 0; n < 4; ++n)
            acc[n] = wmma_bf16(pf, vf[n], acc[n]);
        // rotate pipelined K fragments
        #pragma unroll
        for (int t = 0; t < 2; ++t) {
            #pragma unroll
            for (int c = 0; c < 2; ++c) kf[t][c] = nk[t][c];
        }
    }
    float inv = (l_run > 0.f) ? 1.0f / l_run : 0.f;
    #pragma unroll
    for (int i = 0; i < 8; ++i) {
        float r = __shfl(inv, i + hi * 8, 32);
        int row = q0 + i + hi * 8;
        #pragma unroll
        for (int n = 0; n < 4; ++n) {
            int col = h * DH + n * 16 + lq;
            O[((size_t)b * SS + row) * DD + col] = (__bf16)(acc[n][i] * r);
        }
    }
}

// ---------- host launcher ----------
extern "C" void kernel_launch(void* const* d_in, const int* in_sizes, int n_in,
                              void* d_out, int out_size, void* d_ws, size_t ws_size,
                              hipStream_t stream) {
    const float* x        = (const float*)d_in[0];
    const float* cw       = (const float*)d_in[1];
    const int*   cidx     = (const int*)  d_in[2];
    const float* ewQ      = (const float*)d_in[3];
    const int*   eiQ      = (const int*)  d_in[4];
    const float* ewK      = (const float*)d_in[5];
    const int*   eiK      = (const int*)  d_in[6];
    const float* ewV      = (const float*)d_in[7];
    const int*   eiV      = (const int*)  d_in[8];
    const float* neurons  = (const float*)d_in[9];
    const float* pool     = (const float*)d_in[10];
    const float* W_O      = (const float*)d_in[11];
    float* out            = (float*)d_out;

    char* ws = (char*)d_ws;
    size_t off = 0;
    auto alloc = [&](size_t bytes) { size_t o = off; off = (off + bytes + 255) & ~(size_t)255; return o; };
    __bf16* x_bf  = (__bf16*)(ws + alloc((size_t)BB * SS * DD * 2));
    __bf16* WcT   = (__bf16*)(ws + alloc((size_t)BB * RR * DD * 2));
    __bf16* WqT   = (__bf16*)(ws + alloc((size_t)BB * DD * RR * 2));
    __bf16* WkT   = (__bf16*)(ws + alloc((size_t)BB * DD * RR * 2));
    __bf16* WvT   = (__bf16*)(ws + alloc((size_t)BB * DD * RR * 2));
    __bf16* hbuf  = (__bf16*)(ws + alloc((size_t)BB * SS * RR * 2));
    __bf16* Qb    = (__bf16*)(ws + alloc((size_t)BB * SS * DD * 2));
    __bf16* Kb    = (__bf16*)(ws + alloc((size_t)BB * SS * DD * 2));
    __bf16* VT    = (__bf16*)(ws + alloc((size_t)BB * HH * DH * SS * 2));
    __bf16* Ab    = (__bf16*)(ws + alloc((size_t)BB * SS * DD * 2));
    __bf16* WO_bf = (__bf16*)(ws + alloc((size_t)DD * DD * 2));

    const int T = 256;
    // prep
    cvt_f32_bf16<<<(BB * SS * DD + T - 1) / T, T, 0, stream>>>(x, x_bf, BB * SS * DD);
    cvt_f32_bf16<<<(DD * DD + T - 1) / T, T, 0, stream>>>(W_O, WO_bf, DD * DD);
    build_wct<<<(BB * RR * DD + T - 1) / T, T, 0, stream>>>(cw, cidx, neurons, WcT);
    build_wet<<<(BB * DD * RR + T - 1) / T, T, 0, stream>>>(ewQ, eiQ, pool, WqT);
    build_wet<<<(BB * DD * RR + T - 1) / T, T, 0, stream>>>(ewK, eiK, pool, WkT);
    build_wet<<<(BB * DD * RR + T - 1) / T, T, 0, stream>>>(ewV, eiV, pool, WvT);

    // h = x @ Wc : M=S, N=R, K=D
    gemm_wmma<DD, DD, DD, RR, 0><<<dim3(RR / 64, SS / 256, BB), T, 0, stream>>>(
        x_bf, WcT, hbuf,
        (unsigned long long)SS * DD, (unsigned long long)RR * DD, (unsigned long long)SS * RR);
    // Q/K = h @ W : M=S, N=D, K=R
    gemm_wmma<RR, RR, RR, DD, 0><<<dim3(DD / 64, SS / 256, BB), T, 0, stream>>>(
        hbuf, WqT, Qb,
        (unsigned long long)SS * RR, (unsigned long long)DD * RR, (unsigned long long)SS * DD);
    gemm_wmma<RR, RR, RR, DD, 0><<<dim3(DD / 64, SS / 256, BB), T, 0, stream>>>(
        hbuf, WkT, Kb,
        (unsigned long long)SS * RR, (unsigned long long)DD * RR, (unsigned long long)SS * DD);
    // V stored transposed per (b,h): VT[b][h][dh][s]  (MODE 1, LDC = S)
    gemm_wmma<RR, RR, RR, SS, 1><<<dim3(DD / 64, SS / 256, BB), T, 0, stream>>>(
        hbuf, WvT, VT,
        (unsigned long long)SS * RR, (unsigned long long)DD * RR, (unsigned long long)SS * DD);

    // causal flash attention -> Ab [B,S,D] bf16
    attn_wmma<<<dim3((SS / 16) / 8, HH, BB), T, 0, stream>>>(Qb, Kb, VT, Ab);

    // final: out = Ab @ W_O^T  (Bt = W_O row-major), f32 output
    gemm_wmma<DD, DD, DD, DD, 2><<<dim3(DD / 64, SS / 256, BB), T, 0, stream>>>(
        Ab, WO_bf, out,
        (unsigned long long)SS * DD, 0ULL, (unsigned long long)SS * DD);
}
